// GroupedQueryAttention_2740189135485
// MI455X (gfx1250) — matrix operand order repaired and checked by reading, not compiled, not run on hardware
//
#include <hip/hip_runtime.h>
#include <hip/hip_bf16.h>

typedef __attribute__((ext_vector_type(16))) _Float16 v16h;
typedef __attribute__((ext_vector_type(8)))  float    v8f;

#define Hh    32
#define KVh   16
#define HDd   128
#define Cc    4096
#define MAXS  4096
#define Bb    2
#define NCH   32          // split-S chunks per (b,kv)
#define CHUNK (MAXS/NCH)  // 128 positions per wave
#define SCALE 0.17677669529663687f  // 1/sqrt(H=32)

// ---------------------------------------------------------------------------
// GEMV: Y[b, n] = sum_c X[b,c] * W[c,n], thread computes 4 consecutive n via
// float4 loads. 256 threads/block, grid.x = N/1024, grid.y = B.
// ---------------------------------------------------------------------------
__global__ __launch_bounds__(256) void gemv4(const float* __restrict__ X,
                                             const float* __restrict__ W,
                                             float* __restrict__ Y,
                                             int Cin, int N) {
  const int b = blockIdx.y;
  const float* x = X + (size_t)b * Cin;
  float* y = Y + (size_t)b * N;
  const int col4 = blockIdx.x * 256 + threadIdx.x;   // which group of 4 cols
  const int stride4 = N >> 2;

  __shared__ float xs[Cc];
  for (int i = threadIdx.x; i < Cin; i += 256) xs[i] = x[i];
  __syncthreads();

  const float4* W4 = (const float4*)W;
  float4 acc = make_float4(0.f, 0.f, 0.f, 0.f);
  #pragma unroll 4
  for (int c = 0; c < Cin; ++c) {
    const float xv = xs[c];
    const float4 w = W4[(size_t)c * stride4 + col4];
    acc.x += xv * w.x; acc.y += xv * w.y;
    acc.z += xv * w.z; acc.w += xv * w.w;
  }
  ((float4*)y)[col4] = acc;
}

// ---------------------------------------------------------------------------
// RoPE: grid (nheads, B), 64 threads = 64 (even,odd) pairs of HD=128.
// ---------------------------------------------------------------------------
__global__ __launch_bounds__(64) void rope_kernel(const float* __restrict__ in,
                                                  float* __restrict__ out,
                                                  int nheads,
                                                  const int* __restrict__ spp) {
  const int b = blockIdx.y, h = blockIdx.x, i = threadIdx.x;
  const float inv = powf(10000.f, -((float)(2 * i)) / 128.f);
  const float ang = (float)(*spp) * inv;
  const float cs = cosf(ang), sn = sinf(ang);
  const size_t idx = ((size_t)b * nheads + h) * HDd + 2 * i;
  const float e = in[idx], o = in[idx + 1];
  out[idx]     = e * cs - o * sn;
  out[idx + 1] = e * sn + o * cs;
}

// ---------------------------------------------------------------------------
// Flash-decode partial, one wave per (chunk, kv, b).
// M rows 0..1 = the G=2 query heads sharing kv-head `kv` (h = g*16+kv),
// rows 2..15 zero-padded. QK^T and P*V via v_wmma_f32_16x16x32_f16.
// Writes partial (m, l, o[2][128]) per chunk.
// ---------------------------------------------------------------------------
__global__ __launch_bounds__(32) void attn_partial(
    const float* __restrict__ qr,   // [B, H, HD] roped Q
    const float* __restrict__ krn,  // [B, KV, HD] roped new K
    const float* __restrict__ vnw,  // [B, KV, HD] new V
    const float* __restrict__ kc,   // [B, MAXS, KV, HD]
    const float* __restrict__ vc,   // [B, MAXS, KV, HD]
    const int* __restrict__ spp,
    float* __restrict__ part)       // [B*KV, NCH, 2*130]
{
  const int ch = blockIdx.x;
  const int kv = blockIdx.y;
  const int b  = blockIdx.z;
  const int lane = threadIdx.x;
  const int nl   = lane & 15;     // N / M index within half
  const int half = lane >> 4;
  const int sp   = *spp;

  __shared__ _Float16 pbuf[16][34];   // P tile staging (transpose D->A layout)

  // --- Build Q A-fragments (16x128 f16, rows >= 2 zero), ISA A-layout ---
  v16h aq[4];
  #pragma unroll
  for (int kb = 0; kb < 4; ++kb) {
    #pragma unroll
    for (int j = 0; j < 16; ++j) {
      const int r = j >> 1, odd = j & 1;
      const int k = kb * 32 + ((r & 4) ? 16 : 0) + half * 8 + (r & 3) * 2 + odd;
      float qv = 0.f;
      if (nl < 2) qv = qr[((size_t)b * Hh + (nl * KVh + kv)) * HDd + k];
      aq[kb][j] = (_Float16)qv;
    }
  }

  const v8f vzero = {};
  v8f acc[8];
  #pragma unroll
  for (int t = 0; t < 8; ++t) acc[t] = vzero;
  float mrow[8], lrow[8];
  #pragma unroll
  for (int v = 0; v < 8; ++v) { mrow[v] = -1e30f; lrow[v] = 0.f; }

  const int s_base = ch * CHUNK;
  for (int it = 0; it < CHUNK / 32; ++it) {
    const int st = s_base + it * 32;

    // --- scores: two 16x16 tiles, K-dim = 128 in 4 steps of 32 ---
    v8f sc[2]; sc[0] = vzero; sc[1] = vzero;
    #pragma unroll
    for (int sub = 0; sub < 2; ++sub) {
      const int s = st + sub * 16 + nl;               // B-matrix column = position
      const float* krow = (s == sp)
          ? (krn + ((size_t)b * KVh + kv) * HDd)
          : (kc + (((size_t)b * MAXS + s) * KVh + kv) * HDd);
      #pragma unroll
      for (int kb = 0; kb < 4; ++kb) {
        const float* kp = krow + kb * 32 + half * 8;
        v16h bk;
        #pragma unroll
        for (int j = 0; j < 16; ++j) bk[j] = (_Float16)kp[j + (j & 8)];
        sc[sub] = __builtin_amdgcn_wmma_f32_16x16x32_f16(
            false, aq[kb], false, bk, (short)0, sc[sub], false, false);
      }
    }

    // --- online softmax in C/D layout (lane: col=nl, rows v + 8*half) ---
    float p0[8], p1[8], rmax[8];
    #pragma unroll
    for (int v = 0; v < 8; ++v) {
      float x0 = sc[0][v] * SCALE, x1 = sc[1][v] * SCALE;
      if (st + nl > sp)      x0 = -1e30f;
      if (st + 16 + nl > sp) x1 = -1e30f;
      p0[v] = x0; p1[v] = x1;
      rmax[v] = fmaxf(x0, x1);
    }
    #pragma unroll
    for (int off = 1; off < 16; off <<= 1)
      #pragma unroll
      for (int v = 0; v < 8; ++v)
        rmax[v] = fmaxf(rmax[v], __shfl_xor(rmax[v], off, 32));

    float sf[8], rsum[8];
    #pragma unroll
    for (int v = 0; v < 8; ++v) {
      const float mn = fmaxf(mrow[v], rmax[v]);
      sf[v] = __expf(mrow[v] - mn);
      mrow[v] = mn;
      p0[v] = __expf(p0[v] - mn);
      p1[v] = __expf(p1[v] - mn);
      rsum[v] = p0[v] + p1[v];
    }
    #pragma unroll
    for (int off = 1; off < 16; off <<= 1)
      #pragma unroll
      for (int v = 0; v < 8; ++v)
        rsum[v] += __shfl_xor(rsum[v], off, 32);
    #pragma unroll
    for (int v = 0; v < 8; ++v) lrow[v] = lrow[v] * sf[v] + rsum[v];
    #pragma unroll
    for (int t = 0; t < 8; ++t)
      #pragma unroll
      for (int v = 0; v < 8; ++v) acc[t][v] *= sf[v];

    // --- transpose P (D-layout) -> A-layout via per-wave LDS ---
    #pragma unroll
    for (int v = 0; v < 8; ++v) {
      const int row = v + 8 * half;
      pbuf[row][nl]      = (_Float16)p0[v];
      pbuf[row][16 + nl] = (_Float16)p1[v];
    }
    asm volatile("" ::: "memory");   // keep LDS store->load order (in-order DS)
    v16h pa;
    #pragma unroll
    for (int j = 0; j < 16; ++j) {
      const int r = j >> 1, odd = j & 1;
      const int k = ((r & 4) ? 16 : 0) + half * 8 + (r & 3) * 2 + odd;
      pa[j] = pbuf[nl][k];
    }
    asm volatile("" ::: "memory");

    // --- out += P(16x32) x V(32x128) as 8 N-tiles ---
    #pragma unroll
    for (int t = 0; t < 8; ++t) {
      v16h bv;
      #pragma unroll
      for (int j = 0; j < 16; ++j) {
        const int s = st + half * 8 + j + (j & 8);
        const float* vrow = (s == sp)
            ? (vnw + ((size_t)b * KVh + kv) * HDd)
            : (vc + (((size_t)b * MAXS + s) * KVh + kv) * HDd);
        bv[j] = (_Float16)vrow[t * 16 + nl];
      }
      acc[t] = __builtin_amdgcn_wmma_f32_16x16x32_f16(
          false, pa, false, bv, (short)0, acc[t], false, false);
    }
  }

  // --- write partial (rows 0..1 live in v=0..1 of the half==0 lanes) ---
  float* base = part + ((size_t)(b * KVh + kv) * NCH + ch) * 260;
  if (half == 0) {
    #pragma unroll
    for (int mm = 0; mm < 2; ++mm) {
      if (lane == 0) { base[mm * 130] = mrow[mm]; base[mm * 130 + 1] = lrow[mm]; }
      #pragma unroll
      for (int t = 0; t < 8; ++t)
        base[mm * 130 + 2 + t * 16 + nl] = acc[t][mm];
    }
  }
}

// ---------------------------------------------------------------------------
// Combine split-S partials: grid = B*KV blocks, 256 threads (2 rows x 128 d)
// ---------------------------------------------------------------------------
__global__ __launch_bounds__(256) void attn_combine(const float* __restrict__ part,
                                                    float* __restrict__ attn_out) {
  const int bk = blockIdx.x;          // b*16 + kv
  const int b = bk >> 4, kv = bk & 15;
  const int mm = threadIdx.x >> 7, d = threadIdx.x & 127;
  const float* base = part + (size_t)bk * NCH * 260;

  float mstar = -1e30f;
  for (int ch = 0; ch < NCH; ++ch)
    mstar = fmaxf(mstar, base[ch * 260 + mm * 130]);

  float osum = 0.f, lsum = 0.f;
  for (int ch = 0; ch < NCH; ++ch) {
    const float* p = base + ch * 260 + mm * 130;
    const float w = __expf(p[0] - mstar);
    lsum += w * p[1];
    osum += w * p[2 + d];
  }
  attn_out[((size_t)b * Hh + (mm * KVh + kv)) * HDd + d] = osum / lsum;
}

// ---------------------------------------------------------------------------
extern "C" void kernel_launch(void* const* d_in, const int* in_sizes, int n_in,
                              void* d_out, int out_size, void* d_ws, size_t ws_size,
                              hipStream_t stream) {
  const float* x  = (const float*)d_in[0];
  const float* Wq = (const float*)d_in[1];
  const float* Wk = (const float*)d_in[2];
  const float* Wv = (const float*)d_in[3];
  const float* Wo = (const float*)d_in[4];
  const float* kc = (const float*)d_in[5];
  const float* vc = (const float*)d_in[6];
  const int*   sp = (const int*)d_in[7];

  float* ws     = (float*)d_ws;
  float* q_raw  = ws;               // 8192
  float* k_raw  = q_raw  + 8192;    // 4096
  float* v_new  = k_raw  + 4096;    // 4096
  float* q_rope = v_new  + 4096;    // 8192
  float* k_rope = q_rope + 8192;    // 4096
  float* attn_o = k_rope + 4096;    // 8192
  float* part   = attn_o + 8192;    // B*KV*NCH*260 = 266240

  // QKV projections (bandwidth-bound GEMVs)
  gemv4<<<dim3(4, Bb), 256, 0, stream>>>(x, Wq, q_raw, Cc, Hh * HDd);
  gemv4<<<dim3(2, Bb), 256, 0, stream>>>(x, Wk, k_raw, Cc, KVh * HDd);
  gemv4<<<dim3(2, Bb), 256, 0, stream>>>(x, Wv, v_new, Cc, KVh * HDd);

  // RoPE on q and k
  rope_kernel<<<dim3(Hh,  Bb), 64, 0, stream>>>(q_raw, q_rope, Hh,  sp);
  rope_kernel<<<dim3(KVh, Bb), 64, 0, stream>>>(k_raw, k_rope, KVh, sp);

  // Flash-decode attention with WMMA, split along S
  attn_partial<<<dim3(NCH, KVh, Bb), 32, 0, stream>>>(q_rope, k_rope, v_new,
                                                      kc, vc, sp, part);
  attn_combine<<<Bb * KVh, 256, 0, stream>>>(part, attn_o);

  // Output projection
  gemv4<<<dim3(4, Bb), 256, 0, stream>>>(attn_o, Wo, (float*)d_out, Cc, Cc);
}
